// MSA_20916490732184
// MI455X (gfx1250) — matrix operand order repaired and checked by reading, not compiled, not run on hardware
//
#include <hip/hip_runtime.h>

typedef unsigned short u16;
typedef unsigned int   u32;
typedef __attribute__((ext_vector_type(16))) __bf16 v16bf;
typedef __attribute__((ext_vector_type(8)))  __bf16 v8bf;
typedef __attribute__((ext_vector_type(8)))  float  v8f;

#define HEADS 12
#define DHEAD 64
#define NSEQ  4096
#define DIM   768
#define NTOK  8192

__device__ __forceinline__ u16 f2bf(float f) {
  u32 u = __builtin_bit_cast(u32, f);
  u32 r = u + 0x7FFFu + ((u >> 16) & 1u);   // round-to-nearest-even
  return (u16)(r >> 16);
}

__device__ __forceinline__ v16bf cat8(v8bf lo, v8bf hi) {
  return __builtin_shufflevector(lo, hi, 0,1,2,3,4,5,6,7,8,9,10,11,12,13,14,15);
}

// A fragment: 16 x 32 bf16 tile, row-major [m][k] in LDS with row stride `ld`.
// lanes 0-15 -> row=lane, K={kk+0..7, kk+16..23}; lanes 16-31 -> K shifted by 8.
__device__ __forceinline__ v16bf load_afrag(const u16* base, int ld, int kk, int l, int hi) {
  const u16* p = base + l * ld + kk + hi * 8;
  v8bf a0 = *reinterpret_cast<const v8bf*>(p);
  v8bf a1 = *reinterpret_cast<const v8bf*>(p + 16);
  return cat8(a0, a1);
}

// B fragment: 32 x 16 bf16 (KxN), stored [n][k] row-major with row stride `ld`.
// lanes 0-15 -> col=lane, K={kk+0..7, kk+16..23}; lanes 16-31 -> K shifted by 8.
__device__ __forceinline__ v16bf load_bfrag(const u16* base, int ld, int kk, int l, int hi) {
  const u16* p = base + l * ld + kk + hi * 8;
  v8bf b0 = *reinterpret_cast<const v8bf*>(p);
  v8bf b1 = *reinterpret_cast<const v8bf*>(p + 16);
  return cat8(b0, b1);
}

__device__ __forceinline__ v8f wmma_bf16(v16bf a, v16bf b, v8f c) {
  return __builtin_amdgcn_wmma_f32_16x16x32_bf16(false, a, false, b, (short)0, c, false, false);
}

// Async global -> LDS, 16 bytes per lane; tracked by ASYNCcnt (CDNA5 path).
__device__ __forceinline__ void async_b128(void* lds, const void* g) {
  u32 loff = (u32)(unsigned long long)(uintptr_t)lds;
  asm volatile("global_load_async_to_lds_b128 %0, %1, off" :: "v"(loff), "v"(g) : "memory");
}
__device__ __forceinline__ void wait_async0() { asm volatile("s_wait_asynccnt 0x0" ::: "memory"); }
__device__ __forceinline__ void wait_ds0()    { asm volatile("s_wait_dscnt 0x0"    ::: "memory"); }

// ---------------------------------------------------------------- convert
__global__ __launch_bounds__(256)
void cvt_f32_bf16(const float* __restrict__ s, u16* __restrict__ d, long long n) {
  long long i = ((long long)blockIdx.x * 256 + threadIdx.x) * 4;
  if (i + 3 < n) {
    float4 f = *(const float4*)(s + i);
    ushort4 o;
    o.x = f2bf(f.x); o.y = f2bf(f.y); o.z = f2bf(f.z); o.w = f2bf(f.w);
    *(ushort4*)(d + i) = o;
  }
}

// ---------------------------------------------------------------- V transpose (per head)
// V (tokens x 768 bf16, head-interleaved) -> Vt [(b*12+h)*64 + d][4096]
__global__ __launch_bounds__(256)
void transpose_v(const u16* __restrict__ V, u16* __restrict__ Vt) {
  __shared__ __align__(16) u16 lsT[64][72];
  const int tid = threadIdx.x;
  const int tok0 = blockIdx.x * 64;
  const int h = blockIdx.y;
  const int b = tok0 / NSEQ;
  const int tloc = tok0 % NSEQ;

  #pragma unroll
  for (int i = 0; i < 2; i++) {
    int c = tid + i * 256; int r = c >> 3, q = c & 7;
    *(uint4*)&lsT[r][q * 8] = *(const uint4*)(V + (size_t)(tok0 + r) * DIM + h * DHEAD + q * 8);
  }
  __syncthreads();
  const size_t obase = ((size_t)(b * HEADS + h) * DHEAD) * NSEQ + tloc;
  #pragma unroll
  for (int i = 0; i < 2; i++) {
    int c = tid + i * 256; int d = c >> 3, q = c & 7;
    u16 tmp[8];
    #pragma unroll
    for (int j = 0; j < 8; j++) tmp[j] = lsT[q * 8 + j][d];
    *(uint4*)(Vt + obase + (size_t)d * NSEQ + q * 8) = *(uint4*)tmp;
  }
}

// ---------------------------------------------------------------- GEMM: C = A(MxK) * W(NxK)^T
// 128x256 block, 8 waves (2x4), each wave 64x64 -> 16 WMMA per 16 ds_load_b128.
#define BM 128
#define BN 256
#define BK 32

__global__ __launch_bounds__(256)
void gemm_bf16_wmma(const u16* __restrict__ A, const u16* __restrict__ W,
                    int M, int N, int K,
                    u16* __restrict__ outBf, float outScale,
                    float* __restrict__ outF, const float* __restrict__ bias)
{
  __shared__ __align__(16) u16 lsA[2][BM * BK];   // 2 x 8 KB
  __shared__ __align__(16) u16 lsB[2][BN * BK];   // 2 x 16 KB
  const int tid  = threadIdx.x;
  const int wave = tid >> 5, lane = tid & 31;
  const int l = lane & 15, hi = lane >> 4;
  const int wm = wave >> 2, wn = wave & 3;        // 2 x 4 wave grid
  const int m0 = blockIdx.x * BM, n0 = blockIdx.y * BN;

  auto stage = [&](int k0, int buf) {
    #pragma unroll
    for (int i = 0; i < 2; i++) {                 // A: 512 x 16B chunks
      int c = tid + i * 256; int r = c >> 2, q = c & 3;
      async_b128(&lsA[buf][r * BK + q * 8], A + (size_t)(m0 + r) * K + k0 + q * 8);
    }
    #pragma unroll
    for (int i = 0; i < 4; i++) {                 // B: 1024 x 16B chunks
      int c = tid + i * 256; int r = c >> 2, q = c & 3;
      async_b128(&lsB[buf][r * BK + q * 8], W + (size_t)(n0 + r) * K + k0 + q * 8);
    }
  };

  const v8f vzero = {0.f, 0.f, 0.f, 0.f, 0.f, 0.f, 0.f, 0.f};
  v8f acc[4][4];
  #pragma unroll
  for (int i = 0; i < 4; i++)
    #pragma unroll
    for (int j = 0; j < 4; j++) acc[i][j] = vzero;

  stage(0, 0);
  wait_async0();
  __syncthreads();

  const int nk = K / BK;                           // 24
  for (int kt = 0; kt < nk; kt++) {
    const int cur = kt & 1;
    if (kt + 1 < nk) stage((kt + 1) * BK, cur ^ 1);   // overlap copy with compute

    v16bf af[4], bfr[4];
    #pragma unroll
    for (int tm = 0; tm < 4; tm++)
      af[tm] = load_afrag(&lsA[cur][(wm * 64 + tm * 16) * BK], BK, 0, l, hi);
    #pragma unroll
    for (int tn = 0; tn < 4; tn++)
      bfr[tn] = load_bfrag(&lsB[cur][(wn * 64 + tn * 16) * BK], BK, 0, l, hi);
    #pragma unroll
    for (int tm = 0; tm < 4; tm++)
      #pragma unroll
      for (int tn = 0; tn < 4; tn++)
        acc[tm][tn] = wmma_bf16(af[tm], bfr[tn], acc[tm][tn]);

    wait_ds0();
    wait_async0();
    __syncthreads();
  }

  #pragma unroll
  for (int tm = 0; tm < 4; tm++)
    #pragma unroll
    for (int tn = 0; tn < 4; tn++)
      #pragma unroll
      for (int r = 0; r < 8; r++) {
        int row = m0 + wm * 64 + tm * 16 + r + hi * 8;
        int col = n0 + wn * 64 + tn * 16 + l;
        float v = acc[tm][tn][r];
        size_t idx = (size_t)row * N + col;
        if (outF) outF[idx] = v + (bias ? bias[col] : 0.f);
        else      outBf[idx] = f2bf(v * outScale);
      }
}

// ---------------------------------------------------------------- flash attention (transposed-S form)
// S^T = K * Q^T  -> kv on (VGPR,half), q on lanes: per-lane softmax stats,
// P^T lands B-layout-ready, O^T = V^T * P^T, packed b128 LDS/global stores.
#define QB 128   // q rows per block
#define KB 64    // kv rows per step

__global__ __launch_bounds__(256)
void flash_attn_wmma(const u16* __restrict__ Q, const u16* __restrict__ Km,
                     const u16* __restrict__ Vt, u16* __restrict__ O)
{
  __shared__ __align__(16) u16 lsQ[QB * DHEAD];         // 16 KB
  __shared__ __align__(16) u16 lsK[2][KB * DHEAD];      // 2 x 8 KB
  __shared__ __align__(16) u16 lsVT[2][DHEAD * KB];     // 2 x 8 KB  ([d][kv])
  __shared__ __align__(16) u16 lsP[8 * 16 * KB];        // 16 KB ([q][kv] per wave)

  const int tid = threadIdx.x, wave = tid >> 5, lane = tid & 31;
  const int l = lane & 15, hi = lane >> 4;
  const int q0 = blockIdx.x * QB;
  const int b = blockIdx.y / HEADS, h = blockIdx.y % HEADS;
  const size_t tokBase = (size_t)b * NSEQ;
  const size_t vtBase  = ((size_t)(b * HEADS + h) * DHEAD) * NSEQ;

  auto stage_kv = [&](int kv0, int buf) {
    #pragma unroll
    for (int i = 0; i < 2; i++) {                        // 512 chunks each tile
      int c = tid + i * 256; int r = c >> 3, q = c & 7;
      async_b128(&lsK[buf][r * DHEAD + q * 8],
                 Km + (tokBase + kv0 + r) * DIM + h * DHEAD + q * 8);
      async_b128(&lsVT[buf][r * KB + q * 8],             // r = d row here
                 Vt + vtBase + (size_t)r * NSEQ + kv0 + q * 8);
    }
  };

  #pragma unroll
  for (int i = 0; i < 4; i++) {
    int c = tid + i * 256; int r = c >> 3, q = c & 7;
    async_b128(&lsQ[r * DHEAD + q * 8],
               Q + (tokBase + q0 + r) * DIM + h * DHEAD + q * 8);
  }
  stage_kv(0, 0);
  wait_async0();
  __syncthreads();

  const v8f vzero = {0.f, 0.f, 0.f, 0.f, 0.f, 0.f, 0.f, 0.f};
  float rmax = -1e30f, rsum = 0.f;      // per-lane: stats for q = q0 + wave*16 + l
  v8f oacc[4];
  #pragma unroll
  for (int t = 0; t < 4; t++) oacc[t] = vzero;

  u16* pw = lsP + wave * 16 * KB;

  for (int kv0 = 0; kv0 < NSEQ; kv0 += KB) {
    const int cur = (kv0 / KB) & 1;
    if (kv0 + KB < NSEQ) stage_kv(kv0 + KB, cur ^ 1);    // overlap with compute

    // S^T = K * Q^T : tiles over kv strips; Q pre-scaled by 1/sqrt(d)
    v8f s[4];
    #pragma unroll
    for (int t = 0; t < 4; t++) s[t] = vzero;
    #pragma unroll
    for (int kk = 0; kk < DHEAD; kk += 32) {
      v16bf bq = load_bfrag(lsQ + (wave * 16) * DHEAD, DHEAD, kk, l, hi);
      #pragma unroll
      for (int t = 0; t < 4; t++) {
        v16bf ak = load_afrag(&lsK[cur][(t * 16) * DHEAD], DHEAD, kk, l, hi);
        s[t] = wmma_bf16(ak, bq, s[t]);
      }
    }
    // s[t][r] = S[kv = kv0 + t*16 + r + hi*8][q = q0 + wave*16 + l]

    // online softmax: per-lane over kv, one cross-half shuffle
    float m = -1e30f;
    #pragma unroll
    for (int t = 0; t < 4; t++)
      #pragma unroll
      for (int r = 0; r < 8; r++) m = fmaxf(m, s[t][r]);
    m = fmaxf(m, __shfl_xor(m, 16, 32));
    float nm   = fmaxf(rmax, m);
    float corr = __expf(rmax - nm);
    rmax = nm;
    float rows = 0.f;
    #pragma unroll
    for (int t = 0; t < 4; t++)
      #pragma unroll
      for (int r = 0; r < 8; r++) {
        float p = __expf(s[t][r] - nm);
        s[t][r] = p;
        rows += p;
      }
    rows += __shfl_xor(rows, 16, 32);
    rsum = rsum * corr + rows;
    #pragma unroll
    for (int t = 0; t < 4; t++)
      #pragma unroll
      for (int r = 0; r < 8; r++) oacc[t][r] *= corr;

    // P^T -> per-wave LDS [q][kv]: kv consecutive over r -> packed b128 stores
    #pragma unroll
    for (int t = 0; t < 4; t++) {
      u16 tmp[8];
      #pragma unroll
      for (int r = 0; r < 8; r++) tmp[r] = f2bf(s[t][r]);
      *(uint4*)(pw + l * KB + t * 16 + hi * 8) = *(uint4*)tmp;
    }
    wait_ds0();

    // O^T += V^T * P^T : A = V^T [d][kv] tiles, B = P^T [q][kv]
    #pragma unroll
    for (int kk = 0; kk < KB; kk += 32) {
      v16bf bp = load_bfrag(pw, KB, kk, l, hi);
      #pragma unroll
      for (int t = 0; t < 4; t++) {
        v16bf av = load_afrag(&lsVT[cur][(t * 16) * KB], KB, kk, l, hi);
        oacc[t] = wmma_bf16(av, bp, oacc[t]);
      }
    }

    wait_ds0();
    wait_async0();
    __syncthreads();
  }

  // normalize + store: oacc[t][r] = O[q][d = t*16 + r + hi*8], q = lane's row
  const float inv = 1.f / rsum;
  const size_t base = (tokBase + q0 + wave * 16 + l) * DIM + h * DHEAD;
  #pragma unroll
  for (int t = 0; t < 4; t++) {
    u16 tmp[8];
    #pragma unroll
    for (int r = 0; r < 8; r++) tmp[r] = f2bf(oacc[t][r] * inv);
    *(uint4*)(O + base + t * 16 + hi * 8) = *(uint4*)tmp;
  }
}

// ---------------------------------------------------------------- host
extern "C" void kernel_launch(void* const* d_in, const int* in_sizes, int n_in,
                              void* d_out, int out_size, void* d_ws, size_t ws_size,
                              hipStream_t stream) {
  const float* x  = (const float*)d_in[0];
  const float* wq = (const float*)d_in[1];
  const float* wk = (const float*)d_in[2];
  const float* wv = (const float*)d_in[3];
  const float* wo = (const float*)d_in[4];
  const float* bo = (const float*)d_in[5];
  float* out = (float*)d_out;

  const long long szTok = (long long)NTOK * DIM;   // 6,291,456
  const long long szW   = (long long)DIM * DIM;    //   589,824

  u16* Xbf = (u16*)d_ws;
  u16* Wqb = Xbf + szTok;
  u16* Wkb = Wqb + szW;
  u16* Wvb = Wkb + szW;
  u16* Wob = Wvb + szW;
  u16* Qb  = Wob + szW;
  u16* Kb  = Qb + szTok;
  u16* Vb  = Kb + szTok;
  u16* Ob  = Vb + szTok;
  u16* Vtb = Ob + szTok;   // transposed V, head-major; total ~80 MB workspace

  unsigned bTok = (unsigned)((szTok / 4 + 255) / 256);
  unsigned bW   = (unsigned)((szW   / 4 + 255) / 256);
  cvt_f32_bf16<<<bTok, 256, 0, stream>>>(x,  Xbf, szTok);
  cvt_f32_bf16<<<bW,   256, 0, stream>>>(wq, Wqb, szW);
  cvt_f32_bf16<<<bW,   256, 0, stream>>>(wk, Wkb, szW);
  cvt_f32_bf16<<<bW,   256, 0, stream>>>(wv, Wvb, szW);
  cvt_f32_bf16<<<bW,   256, 0, stream>>>(wo, Wob, szW);

  dim3 gg(NTOK / BM, DIM / BN);   // 64 x 3
  const float qscale = 0.125f;    // 1/sqrt(64), folded into Q
  gemm_bf16_wmma<<<gg, 256, 0, stream>>>(Xbf, Wqb, NTOK, DIM, DIM, Qb, qscale, nullptr, nullptr);
  gemm_bf16_wmma<<<gg, 256, 0, stream>>>(Xbf, Wkb, NTOK, DIM, DIM, Kb, 1.0f,   nullptr, nullptr);
  gemm_bf16_wmma<<<gg, 256, 0, stream>>>(Xbf, Wvb, NTOK, DIM, DIM, Vb, 1.0f,   nullptr, nullptr);

  transpose_v<<<dim3(NTOK / 64, HEADS), 256, 0, stream>>>(Vb, Vtb);

  flash_attn_wmma<<<dim3(NSEQ / QB, 2 * HEADS), 256, 0, stream>>>(Qb, Kb, Vtb, Ob);

  gemm_bf16_wmma<<<gg, 256, 0, stream>>>(Ob, Wob, NTOK, DIM, DIM, nullptr, 1.0f, out, bo);
}
